// diffusion_model_smearing_76802605187801
// MI455X (gfx1250) — matrix operand order repaired and checked by reading, not compiled, not run on hardware
//
#include <hip/hip_runtime.h>

constexpr int kP   = 5;
constexpr int kN   = 30000;
constexpr int kM   = 50000;
constexpr int kB   = 256;
constexpr int kT   = 13;
constexpr int kC   = 16;
constexpr int kL   = 3;
constexpr int kED  = 4;
constexpr int kH   = 128;
constexpr int kTD  = 5;
constexpr int kE   = kP * kP;   // 25 edge types

typedef __attribute__((ext_vector_type(16))) __bf16   bf16x16;
typedef __attribute__((ext_vector_type(8)))  float    f32x8;
typedef __attribute__((ext_vector_type(4)))  unsigned u32x4;

__device__ __forceinline__ __bf16 f2bf(float f) {
  unsigned u = __builtin_bit_cast(unsigned, f);
  unsigned r = u + 0x7FFFu + ((u >> 16) & 1u);   // round-to-nearest-even
  unsigned short s = (unsigned short)(r >> 16);
  return __builtin_bit_cast(__bf16, s);
}
__device__ __forceinline__ unsigned pack2bf(float a, float b) {
  unsigned lo = (unsigned)__builtin_bit_cast(unsigned short, f2bf(a));
  unsigned hi = (unsigned)__builtin_bit_cast(unsigned short, f2bf(b));
  return lo | (hi << 16);
}
__device__ __forceinline__ float sigm(float x) { return 1.f / (1.f + __expf(-x)); }

// ---------------------------------------------------------------------------
// Weight pre-pack: W[g] (K x F, f32, row-major) ->
//   Wp[g][col][kk]  (u32 = bf16 pair for K = 2kk, 2kk+1), col-major along K,
//   zero-padded to Fpad x Kpad/2.  Done ONCE per launch; blocks then copy it
//   straight into LDS with b128s (no converts, no guards in the hot loop).
// ---------------------------------------------------------------------------
__global__ void packw_kernel(const float* __restrict__ W, unsigned* __restrict__ Wp,
                             int K, int F)
{
  int g = blockIdx.y;
  int Kpad = (K + 31) & ~31, Fpad = (F + 15) & ~15;
  int kp2 = Kpad >> 1;
  int total = Fpad * kp2;
  const float* Wg = W + (size_t)g * K * F;
  unsigned* Wpg = Wp + (size_t)g * total;
  for (int i = blockIdx.x * blockDim.x + threadIdx.x; i < total;
       i += gridDim.x * blockDim.x) {
    int col = i / kp2, kk = i - col * kp2;
    int k = 2 * kk;
    float v0 = (k     < K && col < F) ? Wg[(size_t)k * F + col]       : 0.f;
    float v1 = (k + 1 < K && col < F) ? Wg[(size_t)(k + 1) * F + col] : 0.f;
    Wpg[i] = pack2bf(v0, v1);
  }
}

// ---------------------------------------------------------------------------
// Batched affine:  Out[g, n, col] = act( X[gin(g), n, :K] @ W[g] + B[g] )
// One wave -> one 16x16 D tile (v_wmma_f32_16x16x32_bf16).
// A-tile (16 nodes x Kpad) staged in LDS as packed bf16 pairs (shared by all
// waves); W read pre-packed from global.  Inner loop: 4x ds_load_b128 + wmma,
// no bounds checks.  +4-u32 row padding kills LDS bank conflicts.
// blockDim.x = 32 * (Fpad/16);  grid = (N/16, 1, G).
// ACT: 0 = none, 1 = lrelu(0.01), 2 = sigmoid
// ---------------------------------------------------------------------------
template <int ACT>
__global__ __launch_bounds__(256) void affine_wmma(
    const float* __restrict__ X, int xStride,
    const unsigned* __restrict__ Wp, const float* __restrict__ Bias,
    float* __restrict__ Out, int outStride, int outColOff,
    int K, int F, int gmap)
{
  __shared__ unsigned wpk[128 * 84];   // Fpad * (kp2+4) max
  __shared__ unsigned apk[16 * 84];    // 16  * (kp2+4) max
  const int g    = blockIdx.z;
  const int gin  = (gmap == 1) ? (g / kP) : (gmap == 2) ? (g % kP) : g;
  const int Kpad = (K + 31) & ~31;
  const int Fpad = (F + 15) & ~15;
  const int kp2  = Kpad >> 1;
  const int str  = kp2 + 4;            // padded row stride (u32)

  // --- stage packed W[g] into LDS (pure b128 copies, column re-strided) ---
  {
    const u32x4* src = (const u32x4*)(Wp + (size_t)g * Fpad * kp2);
    int kp2_4 = kp2 >> 2, str_4 = str >> 2;
    u32x4* dst = (u32x4*)wpk;
    for (int i = threadIdx.x; i < Fpad * kp2_4; i += blockDim.x) {
      int col = i / kp2_4, j = i - col * kp2_4;
      dst[col * str_4 + j] = src[i];
    }
  }
  // --- stage A tile (16 nodes x Kpad) as bf16, guard K at stage time ---
  {
    __bf16* ah = (__bf16*)apk;
    const int rowbase = blockIdx.x * 16;
    for (int i = threadIdx.x; i < 16 * Kpad; i += blockDim.x) {
      int row = i / Kpad, k = i - row * Kpad;
      float v = (k < K)
        ? X[((size_t)gin * kN + rowbase + row) * xStride + k] : 0.f;
      ah[row * (2 * str) + k] = f2bf(v);
    }
  }
  __syncthreads();

  const int lane = threadIdx.x & 31;
  const int wave = threadIdx.x >> 5;
  const int hf   = lane >> 4;
  const int l16  = lane & 15;
  const int colb = wave * 16;

  const int abase = l16 * str + 8 * hf;            // A: pairs k0/2+8hf+v (v=0..7)
  const int bbase = (colb + l16) * str + 8 * hf;   // B: pairs k0/2+8hf+v (v=0..7)

  f32x8 acc = {0.f, 0.f, 0.f, 0.f, 0.f, 0.f, 0.f, 0.f};
  for (int kh = 0; kh < kp2; kh += 16) {           // 32 K per step
    union { bf16x16 v; u32x4 q[2]; } ua, ub;
    ua.q[0] = *(const u32x4*)&apk[abase + kh];
    ua.q[1] = *(const u32x4*)&apk[abase + kh + 4];
    ub.q[0] = *(const u32x4*)&wpk[bbase + kh];
    ub.q[1] = *(const u32x4*)&wpk[bbase + kh + 4];
    acc = __builtin_amdgcn_wmma_f32_16x16x32_bf16(false, ua.v, false, ub.v,
                                                  (short)0, acc, false, false);
  }

  // D layout: VGPR r, lanes 0-15 -> M=r, lanes 16-31 -> M=8+r ; N = lane%16
#pragma unroll
  for (int r = 0; r < 8; ++r) {
    int m   = r + 8 * hf;
    int col = colb + l16;
    if (col < F) {
      float v = acc[r];
      if (Bias) v += Bias[(size_t)g * F + col];
      if (ACT == 1) v = (v >= 0.f) ? v : 0.01f * v;
      else if (ACT == 2) v = sigm(v);
      Out[((size_t)g * kN + blockIdx.x * 16 + m) * outStride + outColOff + col] = v;
    }
  }
}

// ---------------------------------------------------------------------------
// small helper kernels
// ---------------------------------------------------------------------------
__global__ void temb_kernel(const float* __restrict__ ts,
                            const float* __restrict__ w1, const float* __restrict__ b1,
                            const float* __restrict__ w2, const float* __restrict__ b2,
                            float* __restrict__ temb)
{
  int i = blockIdx.x * blockDim.x + threadIdx.x;
  if (i >= kB) return;
  float t = ts[i];
  float a[kTD];
  for (int j = 0; j < kTD; ++j) {
    float z = t * w1[j] + b1[j];
    a[j] = z * sigm(z);                       // silu
  }
  for (int k = 0; k < kTD; ++k) {
    float s = b2[k];
    for (int j = 0; j < kTD; ++j) s += a[j] * w2[j * kTD + k];
    temb[i * kTD + k] = s;
  }
}

__global__ void prep_kernel(const float* __restrict__ x, const float* __restrict__ cx,
                            float* __restrict__ cat16, float* __restrict__ cond16)
{
  int i = blockIdx.x * blockDim.x + threadIdx.x;    // p*N + n
  if (i >= kP * kN) return;
  const float* xr = x  + (size_t)i * kT;
  const float* cr = cx + (size_t)i * (kC + kL);
  float* a = cat16  + (size_t)i * 16;
  float* c = cond16 + (size_t)i * 16;
  for (int k = 0; k < kT; ++k) a[k] = xr[k];
  for (int k = 0; k < kL; ++k) a[kT + k] = cr[kC + k];
  for (int k = 0; k < kC; ++k) c[k] = cr[k];
}

__global__ void nt_kernel(const int* __restrict__ bidx, const float* __restrict__ temb,
                          float* __restrict__ h133)
{
  int i = blockIdx.x * blockDim.x + threadIdx.x;
  if (i >= kP * kN) return;
  int b = bidx[i];
  float* row = h133 + (size_t)i * 133;
  for (int k = 0; k < kTD; ++k) {
    float v = temb[b * kTD + k];
    row[64 + k]  = v;      // layer-0 input slot (K = 69)
    row[128 + k] = v;      // layer-1/2 + output slot (K = 133)
  }
}

__global__ void mul_kernel(float* __restrict__ h, const float* __restrict__ gate, int n)
{
  int i = blockIdx.x * blockDim.x + threadIdx.x;
  if (i < n) h[i] *= gate[i];
}

__global__ void gate_kernel(const float* __restrict__ hC, const float* __restrict__ sg,
                            float* __restrict__ h133)
{
  int i = blockIdx.x * blockDim.x + threadIdx.x;
  if (i >= kP * kN * kH) return;
  int col = i % kH, row = i / kH;
  h133[(size_t)row * 133 + col] = hC[i] * (1.f + sg[i]);   // h + sigmoid*h
}

__global__ void zero_kernel(float* __restrict__ p, size_t n)
{
  size_t i = (size_t)blockIdx.x * blockDim.x + threadIdx.x;
  size_t st = (size_t)gridDim.x * blockDim.x;
  for (; i < n; i += st) p[i] = 0.f;
}

// Per edge-type precompute: was = W @ a_s, wad = W @ a_d, eproj = We @ a_e
__global__ void wprep_kernel(const float* __restrict__ gW, const float* __restrict__ gas,
                             const float* __restrict__ gad, const float* __restrict__ gWe,
                             const float* __restrict__ gae,
                             float* __restrict__ was, float* __restrict__ wad,
                             float* __restrict__ eproj)
{
  int e = blockIdx.x, f = threadIdx.x;
  const float* W   = gW  + (size_t)e * kH * kH;
  const float* as_ = gas + (size_t)e * kH;
  const float* ad_ = gad + (size_t)e * kH;
  float s1 = 0.f, s2 = 0.f;
  for (int j = 0; j < kH; ++j) { float w = W[f * kH + j]; s1 += w * as_[j]; s2 += w * ad_[j]; }
  was[e * kH + f] = s1;
  wad[e * kH + f] = s2;
  if (f < kED) {
    const float* We = gWe + (size_t)e * kED * kH;
    const float* ae = gae + (size_t)e * kH;
    float s = 0.f;
    for (int j = 0; j < kH; ++j) s += We[f * kH + j] * ae[j];
    eproj[e * kED + f] = s;
  }
}

__global__ void alsald_kernel(const float* __restrict__ h, const float* __restrict__ was,
                              const float* __restrict__ wad,
                              float* __restrict__ als, float* __restrict__ ald)
{
  int n = blockIdx.x * blockDim.x + threadIdx.x;
  int e = blockIdx.y;
  if (n >= kN) return;
  const float* hs = h + ((size_t)(e / kP) * kN + n) * kH;
  const float* hd = h + ((size_t)(e % kP) * kN + n) * kH;
  const float* wsv = was + e * kH;
  const float* wdv = wad + e * kH;
  float s1 = 0.f, s2 = 0.f;
  for (int k = 0; k < kH; ++k) { s1 += hs[k] * wsv[k]; s2 += hd[k] * wdv[k]; }
  als[e * kN + n] = s1;
  ald[e * kN + n] = s2;
}

// alpha + running segment-max (monotonic unsigned key, atomic_max in L2)
__global__ void edge_alpha_kernel(const int* __restrict__ ei, const float* __restrict__ eattr,
                                  const float* __restrict__ als, const float* __restrict__ ald,
                                  const float* __restrict__ eproj,
                                  float* __restrict__ alphaBuf, unsigned* __restrict__ amax)
{
  int i = blockIdx.x * blockDim.x + threadIdx.x;
  if (i >= kE * kM) return;
  int e = i / kM, m = i - e * kM;
  const int* eie = ei + (size_t)e * 2 * kM;
  int s = eie[m], d = eie[kM + m];
  const float* ea = eattr + ((size_t)e * kM + m) * kED;
  const float* ep = eproj + e * kED;
  float a = als[e * kN + s] + ald[e * kN + d];
  for (int j = 0; j < kED; ++j) a += ea[j] * ep[j];
  a = (a >= 0.f) ? a : 0.2f * a;
  alphaBuf[i] = a;
  unsigned ub  = __builtin_bit_cast(unsigned, a);
  unsigned key = (ub & 0x80000000u) ? ~ub : (ub | 0x80000000u);
  atomicMax(&amax[e * kN + d], key);
}

__global__ void edge_exp_kernel(const int* __restrict__ ei, const float* __restrict__ alphaBuf,
                                const unsigned* __restrict__ amax,
                                float* __restrict__ exBuf, float* __restrict__ den)
{
  int i = blockIdx.x * blockDim.x + threadIdx.x;
  if (i >= kE * kM) return;
  int e = i / kM, m = i - e * kM;
  int d = ei[(size_t)e * 2 * kM + kM + m];
  unsigned key = amax[e * kN + d];
  unsigned ub  = (key & 0x80000000u) ? (key ^ 0x80000000u) : ~key;
  float mx = __builtin_bit_cast(float, ub);
  float ex = __expf(alphaBuf[i] - mx);
  exBuf[i] = ex;
  atomicAdd(&den[e * kN + d], ex);
}

// weighted scatter-add of xl rows: one block per edge, 128 lanes = 128 features
__global__ void edge_agg_kernel(const int* __restrict__ ei, const float* __restrict__ exBuf,
                                const float* __restrict__ den, const float* __restrict__ xl,
                                float* __restrict__ agg)
{
  int m = blockIdx.x, e = blockIdx.y, f = threadIdx.x;
  const int* eie = ei + (size_t)e * 2 * kM;
  int s = eie[m], d = eie[kM + m];
  float w = exBuf[(size_t)e * kM + m] / (den[e * kN + d] + 1e-16f);
  atomicAdd(&agg[((size_t)e * kN + d) * kH + f],
            w * xl[((size_t)e * kN + s) * kH + f]);
}

// comm[d,n,f] = lrelu( max_s ( agg[s*P+d,n,f] + gat_b[s*P+d,f] ) )
__global__ void comm_kernel(const float* __restrict__ agg, const float* __restrict__ gb,
                            float* __restrict__ comm)
{
  int i = blockIdx.x * blockDim.x + threadIdx.x;
  if (i >= kP * kN * kH) return;
  int f = i % kH, pn = i / kH, n = pn % kN, d = pn / kN;
  float mx = -3.4e38f;
  for (int s = 0; s < kP; ++s) {
    int e = s * kP + d;
    float v = agg[((size_t)e * kN + n) * kH + f] + gb[e * kH + f];
    mx = fmaxf(mx, v);
  }
  comm[i] = (mx >= 0.f) ? mx : 0.01f * mx;
}

// ---------------------------------------------------------------------------
extern "C" void kernel_launch(void* const* d_in, const int* in_sizes, int n_in,
                              void* d_out, int out_size, void* d_ws, size_t ws_size,
                              hipStream_t stream)
{
  (void)in_sizes; (void)n_in; (void)out_size; (void)ws_size;

  const float* x      = (const float*)d_in[0];
  const float* cond_x = (const float*)d_in[1];
  const float* eattr  = (const float*)d_in[2];
  const float* tsteps = (const float*)d_in[3];
  const float* te1W = (const float*)d_in[4];  const float* te1b = (const float*)d_in[5];
  const float* te2W = (const float*)d_in[6];  const float* te2b = (const float*)d_in[7];
  const float* embW = (const float*)d_in[8];  const float* embb = (const float*)d_in[9];
  const float* cembW= (const float*)d_in[10]; const float* cembb= (const float*)d_in[11];
  const float* paW  = (const float*)d_in[12]; const float* pab  = (const float*)d_in[13];
  const float* pcW  = (const float*)d_in[14]; const float* pcb  = (const float*)d_in[15];
  const float* sl0W = (const float*)d_in[16]; const float* sl0b = (const float*)d_in[17];
  const float* sl12W= (const float*)d_in[18]; const float* sl12b= (const float*)d_in[19];
  const float* sloW = (const float*)d_in[20]; const float* slob = (const float*)d_in[21];
  const float* gatW = (const float*)d_in[22];
  const float* gatAS= (const float*)d_in[23];
  const float* gatAD= (const float*)d_in[24];
  const float* gatWe= (const float*)d_in[25];
  const float* gatAe= (const float*)d_in[26];
  const float* gatB = (const float*)d_in[27];
  const float* xcW  = (const float*)d_in[28]; const float* xcb = (const float*)d_in[29];
  const float* ccW  = (const float*)d_in[30]; const float* ccb = (const float*)d_in[31];
  const float* caW  = (const float*)d_in[32]; const float* cab = (const float*)d_in[33];
  const int*   bidx = (const int*)d_in[34];
  const int*   eidx = (const int*)d_in[35];
  float* out = (float*)d_out;

  // ---- workspace layout (4-byte units) ----
  float* ws = (float*)d_ws;
  size_t o = 0;
  auto alloc = [&](size_t n) { size_t r = o; o += n; return r; };
  const size_t PN = (size_t)kP * kN;
  size_t f_temb  = alloc((size_t)kB * kTD);
  size_t f_cat16 = alloc(PN * 16);
  size_t f_cond  = alloc(PN * 16);
  size_t f_h133  = alloc(PN * 133);
  size_t f_hA    = alloc(PN * kH);     // embed concat / per-layer hidden
  size_t f_tmpG  = alloc(PN * kH);     // sigmoid-gate scratch
  size_t f_hC    = alloc(PN * kH);     // [xc | cc] concat
  size_t f_comm  = alloc(PN * kH);
  size_t f_xl    = alloc((size_t)kE * kN * kH);
  size_t f_agg   = alloc((size_t)kE * kN * kH);
  size_t f_amax  = alloc((size_t)kE * kN);
  size_t f_den   = alloc((size_t)kE * kN);
  size_t f_als   = alloc((size_t)kE * kN);
  size_t f_ald   = alloc((size_t)kE * kN);
  size_t f_alpha = alloc((size_t)kE * kM);
  size_t f_ex    = alloc((size_t)kE * kM);
  size_t f_was   = alloc((size_t)kE * kH);
  size_t f_wad   = alloc((size_t)kE * kH);
  size_t f_ep    = alloc((size_t)kE * kED);
  // packed bf16 weights (u32 each); per-g sizes = Fpad * Kpad/2
  size_t p_emb  = alloc((size_t)5  * 64  * 16);   // K16  F64
  size_t p_cemb = alloc((size_t)5  * 64  * 16);
  size_t p_pa   = alloc((size_t)5  * 128 * 64);   // K128 F128
  size_t p_pc   = alloc((size_t)5  * 64  * 64);   // K128 F64
  size_t p_sl0  = alloc((size_t)5  * 128 * 48);   // K69  F128
  size_t p_sl12 = alloc((size_t)10 * 128 * 80);   // K133 F128
  size_t p_slo  = alloc((size_t)5  * 16  * 80);   // K133 F13
  size_t p_gat  = alloc((size_t)75 * 128 * 64);   // K128 F128
  size_t p_xc   = alloc((size_t)15 * 64  * 64);   // K128 F64
  size_t p_cc   = alloc((size_t)15 * 64  * 64);
  size_t p_ca   = alloc((size_t)15 * 128 * 64);   // K128 F128

  unsigned* wsu = (unsigned*)d_ws;
  const int TPB = 256;
  const int nodeTiles = kN / 16;                    // 1875
  const int gPN  = (int)((PN + TPB - 1) / TPB);
  const int gPNH = (int)((PN * kH + TPB - 1) / TPB);
  const int gEM  = (kE * kM + TPB - 1) / TPB;

  // ---- pre-pack every weight family to bf16-pair column-major (once) ----
  packw_kernel<<<dim3(4,  5), TPB, 0, stream>>>(embW,  wsu + p_emb,  16,  64);
  packw_kernel<<<dim3(4,  5), TPB, 0, stream>>>(cembW, wsu + p_cemb, 16,  64);
  packw_kernel<<<dim3(32, 5), TPB, 0, stream>>>(paW,   wsu + p_pa,   128, 128);
  packw_kernel<<<dim3(16, 5), TPB, 0, stream>>>(pcW,   wsu + p_pc,   128, 64);
  packw_kernel<<<dim3(24, 5), TPB, 0, stream>>>(sl0W,  wsu + p_sl0,  69,  128);
  packw_kernel<<<dim3(40,10), TPB, 0, stream>>>(sl12W, wsu + p_sl12, 133, 128);
  packw_kernel<<<dim3(5,  5), TPB, 0, stream>>>(sloW,  wsu + p_slo,  133, 13);
  packw_kernel<<<dim3(32,75), TPB, 0, stream>>>(gatW,  wsu + p_gat,  128, 128);
  packw_kernel<<<dim3(16,15), TPB, 0, stream>>>(xcW,   wsu + p_xc,   128, 64);
  packw_kernel<<<dim3(16,15), TPB, 0, stream>>>(ccW,   wsu + p_cc,   128, 64);
  packw_kernel<<<dim3(32,15), TPB, 0, stream>>>(caW,   wsu + p_ca,   128, 128);

  // ---- time embedding + input prep ----
  temb_kernel<<<1, kB, 0, stream>>>(tsteps, te1W, te1b, te2W, te2b, ws + f_temb);
  prep_kernel<<<gPN, TPB, 0, stream>>>(x, cond_x, ws + f_cat16, ws + f_cond);
  nt_kernel<<<gPN, TPB, 0, stream>>>(bidx, ws + f_temb, ws + f_h133);

  // ---- embedding MLP (all WMMA) ----
  affine_wmma<0><<<dim3(nodeTiles, 1, kP), 128, 0, stream>>>(
      ws + f_cat16, 16, wsu + p_emb, embb, ws + f_hA, 128, 0, 16, 64, 0);
  affine_wmma<0><<<dim3(nodeTiles, 1, kP), 128, 0, stream>>>(
      ws + f_cond, 16, wsu + p_cemb, cembb, ws + f_hA, 128, 64, 16, 64, 0);
  affine_wmma<2><<<dim3(nodeTiles, 1, kP), 256, 0, stream>>>(
      ws + f_hA, 128, wsu + p_pa, pab, ws + f_tmpG, 128, 0, 128, 128, 0);
  mul_kernel<<<gPNH, TPB, 0, stream>>>(ws + f_hA, ws + f_tmpG, (int)(PN * kH));
  affine_wmma<0><<<dim3(nodeTiles, 1, kP), 128, 0, stream>>>(
      ws + f_hA, 128, wsu + p_pc, pcb, ws + f_h133, 133, 0, 128, 64, 0);

  // ---- layers ----
  for (int l = 0; l < 3; ++l) {
    const unsigned* Wlp = (l == 0) ? wsu + p_sl0
                                   : wsu + p_sl12 + (size_t)(l - 1) * 5 * 128 * 80;
    const float* bl = (l == 0) ? sl0b : sl12b + (size_t)(l - 1) * kP * kH;
    int Kl = (l == 0) ? 69 : 133;

    // h = lrelu(affine(h_in))
    affine_wmma<1><<<dim3(nodeTiles, 1, kP), 256, 0, stream>>>(
        ws + f_h133, 133, Wlp, bl, ws + f_hA, 128, 0, Kl, 128, 0);

    // GAT precompute (folds xr away: ald = h . (W @ a_d))
    wprep_kernel<<<kE, kH, 0, stream>>>(
        gatW + (size_t)l * kE * kH * kH, gatAS + (size_t)l * kE * kH,
        gatAD + (size_t)l * kE * kH, gatWe + (size_t)l * kE * kED * kH,
        gatAe + (size_t)l * kE * kH, ws + f_was, ws + f_wad, ws + f_ep);
    alsald_kernel<<<dim3((kN + 127) / 128, kE), 128, 0, stream>>>(
        ws + f_hA, ws + f_was, ws + f_wad, ws + f_als, ws + f_ald);

    zero_kernel<<<2048, TPB, 0, stream>>>(ws + f_amax, (size_t)2 * kE * kN); // amax+den
    zero_kernel<<<4096, TPB, 0, stream>>>(ws + f_agg, (size_t)kE * kN * kH);

    // xl[e] = h[e/5] @ gat_W[e]   (25 x N x 128, WMMA)
    affine_wmma<0><<<dim3(nodeTiles, 1, kE), 256, 0, stream>>>(
        ws + f_hA, 128, wsu + p_gat + (size_t)l * kE * 128 * 64, nullptr,
        ws + f_xl, 128, 0, 128, 128, 1);

    // edge softmax + aggregation
    edge_alpha_kernel<<<gEM, TPB, 0, stream>>>(
        eidx, eattr, ws + f_als, ws + f_ald, ws + f_ep,
        ws + f_alpha, (unsigned*)(ws + f_amax));
    edge_exp_kernel<<<gEM, TPB, 0, stream>>>(
        eidx, ws + f_alpha, (const unsigned*)(ws + f_amax), ws + f_ex, ws + f_den);
    edge_agg_kernel<<<dim3(kM, kE), kH, 0, stream>>>(
        eidx, ws + f_ex, ws + f_den, ws + f_xl, ws + f_agg);
    comm_kernel<<<gPNH, TPB, 0, stream>>>(
        ws + f_agg, gatB + (size_t)l * kE * kH, ws + f_comm);

    // h = concat[h@xc, comm@cc];  h = h + sigmoid(h@ca) * h  -> h133[:, :128]
    affine_wmma<0><<<dim3(nodeTiles, 1, kP), 128, 0, stream>>>(
        ws + f_hA, 128, wsu + p_xc + (size_t)l * kP * 64 * 64,
        xcb + (size_t)l * kP * 64, ws + f_hC, 128, 0, 128, 64, 0);
    affine_wmma<0><<<dim3(nodeTiles, 1, kP), 128, 0, stream>>>(
        ws + f_comm, 128, wsu + p_cc + (size_t)l * kP * 64 * 64,
        ccb + (size_t)l * kP * 64, ws + f_hC, 128, 64, 128, 64, 0);
    affine_wmma<2><<<dim3(nodeTiles, 1, kP), 256, 0, stream>>>(
        ws + f_hC, 128, wsu + p_ca + (size_t)l * kP * 128 * 64,
        cab + (size_t)l * kP * kH, ws + f_tmpG, 128, 0, 128, 128, 0);
    gate_kernel<<<gPNH, TPB, 0, stream>>>(ws + f_hC, ws + f_tmpG, ws + f_h133);
  }

  // ---- output head: concat[h, nt] @ slout  (K=133, F=13, one wave/block) ----
  affine_wmma<0><<<dim3(nodeTiles, 1, kP), 32, 0, stream>>>(
      ws + f_h133, 133, wsu + p_slo, slob, out, kT, 0, 133, kT, 0);
}
